// LlamaAttentionSparse_45165876084773
// MI455X (gfx1250) — matrix operand order repaired and checked by reading, not compiled, not run on hardware
//
#include <hip/hip_runtime.h>

typedef __attribute__((ext_vector_type(8)))  __bf16 v8bf;
typedef __attribute__((ext_vector_type(16))) __bf16 v16bf;
typedef __attribute__((ext_vector_type(8)))  float  v8f;

#define S_LEN  2048
#define HID_D  4096
#define NH_D   32
#define HD_D   128
#define HEAVY_D  256
#define RECENT_D 256

__device__ __forceinline__ int lane_id() { return threadIdx.x & 31; }

// ---------------------------------------------------------------------------
// Fragment loaders (CDNA5 WMMA 16x16x32 bf16, wave32).
// A (16x32, MxK): lane L holds row M=L&15; elements 0..7 = K = (L>>4)*8 .. +7,
//                 elements 8..15 = K = (L>>4)*8+16 .. +23.  (ISA 7.12.2)
// B (32x16, KxN): lane L holds col N=L&15; elements 0..7 = K = (L>>4)*8 .. +7,
//                 elements 8..15 = K = (L>>4)*8+16 .. +23 (same as A).
// For C = X @ W^T both fragments are row-contiguous gathers from row-major data.
// ---------------------------------------------------------------------------
__device__ __forceinline__ v16bf load_frag_a(const __bf16* tile, int ld) {
  int l = lane_id();
  const __bf16* p = tile + (size_t)(l & 15) * ld + ((l >> 4) << 3);
  v8bf lo = *(const v8bf*)(p);
  v8bf hi = *(const v8bf*)(p + 16);
  return __builtin_shufflevector(lo, hi, 0,1,2,3,4,5,6,7,8,9,10,11,12,13,14,15);
}

__device__ __forceinline__ v16bf load_frag_b(const __bf16* tile, int ld) {
  int l = lane_id();
  const __bf16* p = tile + (size_t)(l & 15) * ld + ((l >> 4) << 3);
  v8bf lo = *(const v8bf*)(p);
  v8bf hi = *(const v8bf*)(p + 16);
  return __builtin_shufflevector(lo, hi, 0,1,2,3,4,5,6,7,8,9,10,11,12,13,14,15);
}

__device__ __forceinline__ v8f wmma_bf16(v16bf a, v16bf b, v8f c) {
  return __builtin_amdgcn_wmma_f32_16x16x32_bf16(false, a, false, b, (short)0, c,
                                                 false, false);
}

// ---------------------------------------------------------------------------
// f32 -> bf16 convert
// ---------------------------------------------------------------------------
__global__ void cvt_kernel(const float* __restrict__ src, __bf16* __restrict__ dst, int n) {
  int i = blockIdx.x * blockDim.x + threadIdx.x;
  if (i < n) dst[i] = (__bf16)src[i];
}

// ---------------------------------------------------------------------------
// out[M,N] = X[M,K] @ W[N,K]^T   (bf16 in, f32 acc, bf16 or f32 out)
// Block = 256 threads = 8 waves (2x4), wave tile 64x64, block tile 128x256.
// Working set is L2-resident (192MB L2 > 48MB of X+W per GEMM), so fragments
// are loaded straight from global; prefetch the next k-slab.
// ---------------------------------------------------------------------------
template <bool OUT_F32>
__global__ __launch_bounds__(256) void gemm_xwt(const __bf16* __restrict__ X,
                                                const __bf16* __restrict__ W,
                                                void* __restrict__ out,
                                                int M, int N, int K) {
  int wave = threadIdx.x >> 5;
  int wy = wave >> 2, wx = wave & 3;
  int m0 = blockIdx.y * 128 + wy * 64;
  int n0 = blockIdx.x * 256 + wx * 64;
  int l = lane_id();

  v8f zero = {0.f,0.f,0.f,0.f,0.f,0.f,0.f,0.f};
  v8f acc[4][4];
#pragma unroll
  for (int i = 0; i < 4; ++i)
#pragma unroll
    for (int j = 0; j < 4; ++j) acc[i][j] = zero;

  for (int kk = 0; kk < K; kk += 32) {
    if (kk + 32 < K) {
      __builtin_prefetch(X + (size_t)(m0 + (l & 15)) * K + kk + 32, 0, 1);
      __builtin_prefetch(W + (size_t)(n0 + (l & 15)) * K + kk + 32, 0, 1);
    }
    v16bf a[4], b[4];
#pragma unroll
    for (int i = 0; i < 4; ++i)
      a[i] = load_frag_a(X + (size_t)(m0 + 16 * i) * K + kk, K);
#pragma unroll
    for (int j = 0; j < 4; ++j)
      b[j] = load_frag_b(W + (size_t)(n0 + 16 * j) * K + kk, K);
#pragma unroll
    for (int i = 0; i < 4; ++i)
#pragma unroll
      for (int j = 0; j < 4; ++j)
        acc[i][j] = wmma_bf16(a[i], b[j], acc[i][j]);
  }

  int cm = (l >> 4) << 3;  // C layout: lane holds col n=l&15, rows cm..cm+7
  int cn = l & 15;
#pragma unroll
  for (int i = 0; i < 4; ++i)
#pragma unroll
    for (int j = 0; j < 4; ++j)
#pragma unroll
      for (int r = 0; r < 8; ++r) {
        size_t row = (size_t)(m0 + 16 * i + cm + r);
        size_t col = (size_t)(n0 + 16 * j + cn);
        if (OUT_F32)
          ((float*)out)[row * N + col] = acc[i][j][r];
        else
          ((__bf16*)out)[row * N + col] = (__bf16)acc[i][j][r];
      }
}

// ---------------------------------------------------------------------------
// RoPE + layout: qproj/kproj [S,HID] -> qh/kh [NH,S,HD];  vproj -> vt [NH,HD,S]
// ---------------------------------------------------------------------------
__global__ void rope_kernel(const __bf16* __restrict__ qp, const __bf16* __restrict__ kp,
                            const __bf16* __restrict__ vp, const int* __restrict__ pos,
                            __bf16* __restrict__ qh, __bf16* __restrict__ kh,
                            __bf16* __restrict__ vt) {
  int idx = blockIdx.x * blockDim.x + threadIdx.x;
  if (idx >= S_LEN * HID_D) return;
  int s = idx >> 12;      // / HID
  int c = idx & 4095;
  int h = c >> 7;         // / HD
  int d = c & 127;
  int half = d & 63;
  float p = (float)pos[s];
  // inv_freq = 10000^(-half/64) = exp(-ln(10000)*half/64)
  float inv = __expf(-9.210340371976184f * (float)half * (1.0f / 64.0f));
  float sn, cs;
  __sincosf(p * inv, &sn, &cs);
  float q = (float)qp[idx];
  float k = (float)kp[idx];
  int pidx = (d < 64) ? (idx + 64) : (idx - 64);
  float qr = (float)qp[pidx];
  float kr = (float)kp[pidx];
  if (d < 64) { qr = -qr; kr = -kr; }
  size_t o = (size_t)h * S_LEN * HD_D + (size_t)s * HD_D + d;
  qh[o] = (__bf16)(q * cs + qr * sn);
  kh[o] = (__bf16)(k * cs + kr * sn);
  vt[(size_t)h * HD_D * S_LEN + (size_t)d * S_LEN + s] = vp[idx];
}

// ---------------------------------------------------------------------------
// Sparse flash attention. One wave per (head, 16-query tile). Key tiles of 32:
// heavy range [0,256) and recent range [i-256, i], disjoint masks. Online
// softmax with 16-lane butterfly reductions; P goes C-layout -> LDS -> A-frag.
// ---------------------------------------------------------------------------
__global__ __launch_bounds__(256) void attn_sparse(const __bf16* __restrict__ qh,
                                                   const __bf16* __restrict__ kh,
                                                   const __bf16* __restrict__ vt,
                                                   __bf16* __restrict__ outb) {
  __shared__ __align__(32) __bf16 pbuf[8][16 * 32];
  int wave = threadIdx.x >> 5;
  int gw = blockIdx.x * 8 + wave;
  int head = gw >> 7;        // 128 q-tiles per head
  int qt = gw & 127;
  int qbase = qt << 4;
  int l = lane_id();
  int cn = l & 15;
  int cm = (l >> 4) << 3;
  const float scale = 0.08838834764831845f;  // 1/sqrt(128)

  const __bf16* qhh = qh + (size_t)head * S_LEN * HD_D;
  const __bf16* khh = kh + (size_t)head * S_LEN * HD_D;
  const __bf16* vth = vt + (size_t)head * HD_D * S_LEN;
  __bf16* pb = &pbuf[wave][0];

  v16bf aq[4];
#pragma unroll
  for (int d = 0; d < 4; ++d)
    aq[d] = load_frag_a(qhh + (size_t)qbase * HD_D + d * 32, HD_D);

  v8f zero = {0.f,0.f,0.f,0.f,0.f,0.f,0.f,0.f};
  v8f o[8];
  float mrow[8], lrow[8];
#pragma unroll
  for (int t = 0; t < 8; ++t) o[t] = zero;
#pragma unroll
  for (int r = 0; r < 8; ++r) { mrow[r] = -1e30f; lrow[r] = 0.f; }

  auto process = [&](int kb, bool recent) {
    v8f s0 = zero, s1 = zero;
#pragma unroll
    for (int d = 0; d < 4; ++d) {
      v16bf b0 = load_frag_b(khh + (size_t)kb * HD_D + d * 32, HD_D);
      v16bf b1 = load_frag_b(khh + (size_t)(kb + 16) * HD_D + d * 32, HD_D);
      s0 = wmma_bf16(aq[d], b0, s0);
      s1 = wmma_bf16(aq[d], b1, s1);
    }
#pragma unroll
    for (int r = 0; r < 8; ++r) {
      int ig = qbase + cm + r;
      int j0 = kb + cn, j1 = kb + 16 + cn;
      float v0 = s0[r] * scale, v1 = s1[r] * scale;
      bool k0, k1;
      if (recent) {
        k0 = (j0 >= HEAVY_D) && (j0 + RECENT_D >= ig) && (j0 <= ig);
        k1 = (j1 >= HEAVY_D) && (j1 + RECENT_D >= ig) && (j1 <= ig);
      } else {
        k0 = (j0 < HEAVY_D) && (j0 <= ig);
        k1 = (j1 < HEAVY_D) && (j1 <= ig);
      }
      v0 = k0 ? v0 : -1e30f;
      v1 = k1 ? v1 : -1e30f;
      float mx = fmaxf(v0, v1);
#pragma unroll
      for (int off = 1; off < 16; off <<= 1)
        mx = fmaxf(mx, __shfl_xor(mx, off, 32));
      float nm = fmaxf(mrow[r], mx);
      float corr = __expf(mrow[r] - nm);
      float p0 = __expf(v0 - nm);
      float p1 = __expf(v1 - nm);
      float ps = p0 + p1;
#pragma unroll
      for (int off = 1; off < 16; off <<= 1)
        ps += __shfl_xor(ps, off, 32);
      lrow[r] = lrow[r] * corr + ps;
      mrow[r] = nm;
#pragma unroll
      for (int t = 0; t < 8; ++t) o[t][r] = o[t][r] * corr;
      pb[(cm + r) * 32 + cn]      = (__bf16)p0;
      pb[(cm + r) * 32 + 16 + cn] = (__bf16)p1;
    }
    v16bf pf = load_frag_a(pb, 32);  // P as A-frag (16q x 32k), via LDS
#pragma unroll
    for (int t = 0; t < 8; ++t) {
      v16bf bv = load_frag_b(vth + (size_t)(t * 16) * S_LEN + kb, S_LEN);
      o[t] = wmma_bf16(pf, bv, o[t]);
    }
  };

  int heavy_end = (HEAVY_D < qbase + 16) ? HEAVY_D : (qbase + 16);
  for (int kb = 0; kb < heavy_end; kb += 32) process(kb, false);
  int rs = qbase - RECENT_D;
  if (rs < HEAVY_D) rs = HEAVY_D;
  rs &= ~31;
  for (int kb = rs; kb < qbase + 16; kb += 32) process(kb, true);

#pragma unroll
  for (int t = 0; t < 8; ++t)
#pragma unroll
    for (int r = 0; r < 8; ++r) {
      size_t row = (size_t)(qbase + cm + r);
      outb[row * HID_D + (size_t)head * HD_D + t * 16 + cn] =
          (__bf16)(o[t][r] / lrow[r]);
    }
}

// ---------------------------------------------------------------------------
extern "C" void kernel_launch(void* const* d_in, const int* in_sizes, int n_in,
                              void* d_out, int out_size, void* d_ws, size_t ws_size,
                              hipStream_t stream) {
  const float* hs  = (const float*)d_in[0];
  // d_in[1] = attention_mask: standard causal, 0 on kept entries -> not needed
  const int*   pos = (const int*)d_in[2];
  const float* wq  = (const float*)d_in[3];
  const float* wk  = (const float*)d_in[4];
  const float* wv  = (const float*)d_in[5];
  const float* wo  = (const float*)d_in[6];

  const size_t MB = 1ull << 20;
  if (ws_size < 240 * MB) return;
  char* ws = (char*)d_ws;
  __bf16* hsb   = (__bf16*)(ws + 0);         // 16MB (reused as attn_out later)
  __bf16* wqb   = (__bf16*)(ws + 16 * MB);   // 32MB
  __bf16* wkb   = (__bf16*)(ws + 48 * MB);   // 32MB
  __bf16* wvb   = (__bf16*)(ws + 80 * MB);   // 32MB
  __bf16* wob   = (__bf16*)(ws + 112 * MB);  // 32MB
  __bf16* qproj = (__bf16*)(ws + 144 * MB);  // 16MB
  __bf16* kproj = (__bf16*)(ws + 160 * MB);  // 16MB
  __bf16* vproj = (__bf16*)(ws + 176 * MB);  // 16MB
  __bf16* qhb   = (__bf16*)(ws + 192 * MB);  // 16MB
  __bf16* khb   = (__bf16*)(ws + 208 * MB);  // 16MB
  __bf16* vtb   = (__bf16*)(ws + 224 * MB);  // 16MB

  int nHS = S_LEN * HID_D;   // 8.4M
  int nW  = HID_D * HID_D;   // 16.8M
  cvt_kernel<<<(nHS + 255) / 256, 256, 0, stream>>>(hs, hsb, nHS);
  cvt_kernel<<<(nW + 255) / 256, 256, 0, stream>>>(wq, wqb, nW);
  cvt_kernel<<<(nW + 255) / 256, 256, 0, stream>>>(wk, wkb, nW);
  cvt_kernel<<<(nW + 255) / 256, 256, 0, stream>>>(wv, wvb, nW);
  cvt_kernel<<<(nW + 255) / 256, 256, 0, stream>>>(wo, wob, nW);

  dim3 gg(HID_D / 256, S_LEN / 128);  // (16,16)
  gemm_xwt<false><<<gg, 256, 0, stream>>>(hsb, wqb, qproj, S_LEN, HID_D, HID_D);
  gemm_xwt<false><<<gg, 256, 0, stream>>>(hsb, wkb, kproj, S_LEN, HID_D, HID_D);
  gemm_xwt<false><<<gg, 256, 0, stream>>>(hsb, wvb, vproj, S_LEN, HID_D, HID_D);

  rope_kernel<<<(nHS + 255) / 256, 256, 0, stream>>>(qproj, kproj, vproj, pos,
                                                     qhb, khb, vtb);

  __bf16* aout = hsb;  // projections done; reuse hs buffer
  attn_sparse<<<(NH_D * 128) / 8, 256, 0, stream>>>(qhb, khb, vtb, aout);

  gemm_xwt<true><<<gg, 256, 0, stream>>>(aout, wob, d_out, S_LEN, HID_D, HID_D);
}
